// HANLayer_44074954392046
// MI455X (gfx1250) — matrix operand order repaired and checked by reading, not compiled, not run on hardware
//
#include <hip/hip_runtime.h>
#include <hip/hip_bf16.h>
#include <math.h>

#define LSEQ 60
#define BB   8
#define DDIM 512
#define DII  1024
#define DSS  64
#define NROWS (LSEQ*BB)   // 480

typedef __attribute__((ext_vector_type(16))) _Float16 v16h;
typedef __attribute__((ext_vector_type(8)))  float    v8f;

__device__ __forceinline__ void pack8(v16h& v, int o, float4 p, float4 q) {
  v[o+0]=(_Float16)p.x; v[o+1]=(_Float16)p.y; v[o+2]=(_Float16)p.z; v[o+3]=(_Float16)p.w;
  v[o+4]=(_Float16)q.x; v[o+5]=(_Float16)q.y; v[o+6]=(_Float16)q.z; v[o+7]=(_Float16)q.w;
}

// C[M,N] = act( A[M,K] @ W^T + bias ), W is (N,K) row-major, fp32 in/out,
// fp16 WMMA inner product. One wave -> 16 x (NT*16) tile; NT is compile-time,
// so there are NO branches around the WMMAs (EXEC stays all-ones).
// Requires M%16==0, (N/16)%NT==0, K%32==0.
template<int NT>
__global__ void __launch_bounds__(256)
k_gemm_wmma(const float* __restrict__ A, const float* __restrict__ W,
            const float* __restrict__ bias, float* __restrict__ C,
            int M, int N, int K, int act)
{
  int wave = threadIdx.x >> 5;
  int lane = threadIdx.x & 31;
  int tilesN  = N >> 4;
  int groupsN = tilesN / NT;
  long grp   = (long)blockIdx.x * 8 + wave;
  long total = (long)(M >> 4) * groupsN;
  if (grp >= total) return;                  // wave-uniform; execz-skipped
  int mt  = (int)(grp / groupsN);
  int nt0 = (int)(grp % groupsN) * NT;
  int half  = lane >> 4;
  int mrow  = mt*16 + (lane & 15);
  int ncol0 = nt0*16 + (lane & 15);

  v8f acc[NT];
  #pragma unroll
  for (int t = 0; t < NT; ++t) acc[t] = (v8f){};

  for (int k0 = 0; k0 < K; k0 += 32) {
    // A fragment: lane holds row; k = k0 + half*16 + e
    const float* arow = A + (size_t)mrow*K + k0;
    float4 a0 = *(const float4*)(arow + half*16);
    float4 a1 = *(const float4*)(arow + half*16 + 4);
    float4 a2 = *(const float4*)(arow + half*16 + 8);
    float4 a3 = *(const float4*)(arow + half*16 + 12);
    v16h av;
    pack8(av, 0, a0, a1); pack8(av, 8, a2, a3);
    #pragma unroll
    for (int t = 0; t < NT; ++t) {
      // B fragment: lane holds col; k = k0 + half*16 + e
      const float* brow = W + (size_t)(ncol0 + t*16)*K + k0 + half*16;
      float4 b0 = ((const float4*)brow)[0];
      float4 b1 = ((const float4*)brow)[1];
      float4 b2 = ((const float4*)brow)[2];
      float4 b3 = ((const float4*)brow)[3];
      v16h bv;
      pack8(bv, 0, b0, b1); pack8(bv, 8, b2, b3);
      acc[t] = __builtin_amdgcn_wmma_f32_16x16x32_f16(false, av, false, bv,
                                                      (short)0, acc[t], false, false);
    }
  }
  #pragma unroll
  for (int t = 0; t < NT; ++t) {
    int nc = ncol0 + t*16;
    float bval = bias ? bias[nc] : 0.f;
    #pragma unroll
    for (int r = 0; r < 8; ++r) {
      int m = mt*16 + r + half*8;
      float v = acc[t][r] + bval;
      if (act == 1) v = v > 0.f ? v : 0.f;
      C[(size_t)m*N + nc] = v;
    }
  }
}

// out[n*K+k] = in[k*N+n]  (tiled, coalesced both sides)
__global__ void __launch_bounds__(256)
k_transpose_w(const float* __restrict__ in, float* __restrict__ out, int Kd, int Nd)
{
  __shared__ float tile[32][33];
  int kb = blockIdx.x*32, nb = blockIdx.y*32;
  int tx = threadIdx.x & 31, ty = threadIdx.x >> 5;
  for (int i = ty; i < 32; i += 8) {
    int k = kb + i, n = nb + tx;
    tile[i][tx] = (k < Kd && n < Nd) ? in[(size_t)k*Nd + n] : 0.f;
  }
  __syncthreads();
  for (int i = ty; i < 32; i += 8) {
    int n = nb + i, k = kb + tx;
    if (n < Nd && k < Kd) out[(size_t)n*Kd + k] = tile[tx][i];
  }
}

// ---------------- elementwise / small kernels ----------------

__global__ void k_transpose_in(const float* __restrict__ sq, const float* __restrict__ sv,
                               float* __restrict__ Qf, float* __restrict__ Vf)
{
  int idx = blockIdx.x*256 + threadIdx.x;
  if (idx >= NROWS*DDIM) return;
  int d = idx % DDIM; int r = idx / DDIM; int b = r % BB; int t = r / BB;
  size_t s = ((size_t)b*LSEQ + t)*DDIM + d;
  Qf[idx] = sq[s]; Vf[idx] = sv[s];
}

__global__ void k_negexp(const float* __restrict__ in, float* __restrict__ out, int n)
{ int i = blockIdx.x*256 + threadIdx.x; if (i < n) out[i] = -expf(in[i]); }

// causal depthwise conv (window 4) in scan order + SiLU. rev=1: scan runs t=L-1..0
__global__ void k_conv_silu(const float* __restrict__ XZ, const float* __restrict__ cw,
                            const float* __restrict__ cb, float* __restrict__ XC, int rev)
{
  int idx = blockIdx.x*256 + threadIdx.x;
  if (idx >= NROWS*DII) return;
  int di = idx % DII; int rb = idx / DII; int b = rb % BB; int s = rb / BB;
  float acc = cb[di];
  #pragma unroll
  for (int k = 0; k < 4; ++k) {
    int sp = s - 3 + k;
    if (sp >= 0) {
      int t = rev ? (LSEQ-1-sp) : sp;
      acc += XZ[((size_t)(t*BB + b))*(2*DII) + di] * cw[di*4 + k];
    }
  }
  XC[idx] = acc / (1.f + expf(-acc));   // silu
}

// dt = softplus(xdb[:, :32] @ dt_w^T + dt_b)
__global__ void k_dt(const float* __restrict__ XDB, const float* __restrict__ dtw,
                     const float* __restrict__ dtb, float* __restrict__ DT)
{
  int idx = blockIdx.x*256 + threadIdx.x;
  if (idx >= NROWS*DII) return;
  int di = idx % DII; int r = idx / DII;
  float acc = dtb[di];
  const float* xr = XDB + (size_t)r*160;
  #pragma unroll
  for (int j = 0; j < 32; ++j) acc += xr[j] * dtw[di*32 + j];
  DT[idx] = (acc > 20.f) ? acc : log1pf(expf(acc));
}

__global__ void k_rowdot(const float* __restrict__ X, const float* __restrict__ wv,
                         float* __restrict__ out)
{
  int r = blockIdx.x*256 + threadIdx.x;
  if (r >= NROWS) return;
  float acc = 0.f;
  const float* xr = X + (size_t)r*DDIM;
  for (int d = 0; d < DDIM; ++d) acc += xr[d]*wv[d];
  out[r] = acc;
}

// ------------- whole scan in ONE launch -------------------------------------
// 16 workgroups = (dir, batch); 1024 threads; thread tid owns channel di=tid,
// its 64-element SSM state lives in registers for all 60 steps.

struct ScanArgs {
  const float *xconv_a, *dt_a, *xdb_a, *A_a, *Dp_a, *xz_a;   // a-branch mamba
  const float *xconv_v, *dt_v, *xdb_v, *A_v;                 // v-branch mamba
  const float *Ea, *Ev, *eba, *ebv;                          // hoisted fuse projections (time order)
  const float *hp1_w, *hp1_b, *hp2_w, *hp2_b;
  float *Ya;                                                 // (NROWS, DII) scan order
  int rev;
};

__global__ void __launch_bounds__(1024)
k_scan(ScanArgs d0, ScanArgs d1)
{
  ScanArgs d = (blockIdx.x & 1) ? d1 : d0;
  int b   = blockIdx.x >> 1;
  int tid = threadIdx.x;                 // == di
  int lane = tid & 31, wid = tid >> 5;

  __shared__ float h1[DSS], h2[DSS];
  __shared__ float xba[160], xbv[160];
  __shared__ float red1[32], red2[32];
  __shared__ float wsh[2];

  if (tid < DSS) { h1[tid] = d.hp1_w[tid]; h2[tid] = d.hp2_w[tid]; }
  float h1b = d.hp1_b[0], h2b = d.hp2_b[0];
  float DpA = d.Dp_a[tid];
  const float* Aa = d.A_a + (size_t)tid*DSS;
  const float* Av = d.A_v + (size_t)tid*DSS;

  float st[DSS];
  #pragma unroll
  for (int n = 0; n < DSS; ++n) st[n] = 0.f;

  for (int s = 0; s < LSEQ; ++s) {
    int rs = s*BB + b;
    int rt = (d.rev ? (LSEQ-1-s) : s)*BB + b;

    __syncthreads();   // prev-step xba readers done; h1/h2 ready on first iter
    if (tid < 160)      xba[tid]       = d.xdb_a[(size_t)rs*160 + tid];
    else if (tid < 320) xbv[tid - 160] = d.xdb_v[(size_t)rs*160 + (tid - 160)];

    // ---- fuse weights from PRE-update ("shared") state ----
    float u1 = h1b, u2 = h2b;
    #pragma unroll
    for (int n = 0; n < DSS; ++n) { u1 += st[n]*h1[n]; u2 += st[n]*h2[n]; }
    float acc1 = u1 * d.Ea[(size_t)rt*DII + tid];
    float acc2 = u2 * d.Ev[(size_t)rt*DII + tid];
    #pragma unroll
    for (int off = 16; off > 0; off >>= 1) {
      acc1 += __shfl_xor(acc1, off, 32);
      acc2 += __shfl_xor(acc2, off, 32);
    }
    if (lane == 0) { red1[wid] = acc1; red2[wid] = acc2; }
    __syncthreads();
    if (tid < 32) {
      float r1 = red1[tid], r2 = red2[tid];
      #pragma unroll
      for (int off = 16; off > 0; off >>= 1) {
        r1 += __shfl_xor(r1, off, 32);
        r2 += __shfl_xor(r2, off, 32);
      }
      if (tid == 0) {
        float sim_a = expf(r1 + d.eba[rt]) + 1.f;
        float sim_v = expf(r2 + d.ebv[rt]) + 1.f;
        float wa = sim_a / (sim_a + sim_v);
        wsh[0] = wa; wsh[1] = 1.f - wa;
      }
    }
    __syncthreads();

    // ---- SSM update (both branches), y output (a-branch), fused carry ----
    float wa = wsh[0], wv = wsh[1];
    float dta = d.dt_a[(size_t)rs*DII + tid], xa = d.xconv_a[(size_t)rs*DII + tid];
    float dtv = d.dt_v[(size_t)rs*DII + tid], xv = d.xconv_v[(size_t)rs*DII + tid];
    float dxa = dta*xa, dxv = dtv*xv, ysum = 0.f;
    #pragma unroll
    for (int j = 0; j < DSS/4; ++j) {
      float4 a4 = *(const float4*)(Aa + j*4);
      float4 v4 = *(const float4*)(Av + j*4);
      float ae[4] = {a4.x, a4.y, a4.z, a4.w};
      float ve[4] = {v4.x, v4.y, v4.z, v4.w};
      #pragma unroll
      for (int e = 0; e < 4; ++e) {
        int n = j*4 + e;
        float sa = st[n]*expf(dta*ae[e]) + dxa*xba[32 + n];
        float sv = st[n]*expf(dtv*ve[e]) + dxv*xbv[32 + n];
        ysum += sa * xba[96 + n];
        st[n] = wa*sa + wv*sv;
      }
    }
    float y = ysum + DpA*xa;
    float z = d.xz_a[(size_t)rt*(2*DII) + DII + tid];
    y *= z / (1.f + expf(-z));
    d.Ya[(size_t)rs*DII + tid] = y;
  }
}

// ---------------- attention core (per (b,h) block) ---------------------------

__global__ void __launch_bounds__(256)
k_mha(const float* __restrict__ QKV, float* __restrict__ O)
{
  __shared__ float Qs[LSEQ*64], Ks[LSEQ*64], Vs[LSEQ*64], Ss[LSEQ*64];
  int b = blockIdx.x >> 3, h = blockIdx.x & 7;
  int tid = threadIdx.x;
  for (int i = tid; i < LSEQ*64; i += 256) {
    int t = i >> 6, dd = i & 63;
    size_t base = ((size_t)(t*BB + b))*1536 + h*64 + dd;
    Qs[i] = QKV[base]; Ks[i] = QKV[base + 512]; Vs[i] = QKV[base + 1024];
  }
  __syncthreads();
  for (int i = tid; i < LSEQ*LSEQ; i += 256) {
    int qi = i / LSEQ, ki = i % LSEQ;
    float acc = 0.f;
    #pragma unroll 8
    for (int dd = 0; dd < 64; ++dd) acc += Qs[qi*64+dd]*Ks[ki*64+dd];
    Ss[qi*64 + ki] = acc * 0.125f;   // 1/sqrt(64)
  }
  __syncthreads();
  if (tid < LSEQ) {
    float mx = -1e30f;
    for (int k = 0; k < LSEQ; ++k) mx = fmaxf(mx, Ss[tid*64+k]);
    float sum = 0.f;
    for (int k = 0; k < LSEQ; ++k) { float e = expf(Ss[tid*64+k]-mx); Ss[tid*64+k] = e; sum += e; }
    float inv = 1.f/sum;
    for (int k = 0; k < LSEQ; ++k) Ss[tid*64+k] *= inv;
  }
  __syncthreads();
  for (int i = tid; i < LSEQ*64; i += 256) {
    int qi = i >> 6, dd = i & 63;
    float acc = 0.f;
    for (int k = 0; k < LSEQ; ++k) acc += Ss[qi*64+k]*Vs[k*64+dd];
    O[((size_t)(qi*BB + b))*DDIM + h*64 + dd] = acc;
  }
}

// x = 2*q + a_out + reverse(a_f) + src2
__global__ void k_combine(const float* __restrict__ Qf, const float* __restrict__ Oa1,
                          const float* __restrict__ Oa2, const float* __restrict__ S2,
                          float* __restrict__ X1)
{
  int idx = blockIdx.x*256 + threadIdx.x;
  if (idx >= NROWS*DDIM) return;
  int dd = idx % DDIM; int r = idx / DDIM; int b = r % BB; int t = r / BB;
  size_t r2 = (size_t)((LSEQ-1-t)*BB + b);
  X1[idx] = 2.f*Qf[idx] + Oa1[idx] + Oa2[r2*DDIM + dd] + S2[idx];
}

// LayerNorm over D=512; optional residual add; remap=1 writes (B,L,D) order.
__global__ void __launch_bounds__(256)
k_ln(const float* __restrict__ X, const float* __restrict__ R,
     const float* __restrict__ g, const float* __restrict__ bt,
     float* __restrict__ out, int remap)
{
  int r = blockIdx.x, tid = threadIdx.x;
  const float* xr = X + (size_t)r*DDIM;
  float v0 = xr[tid]       + (R ? R[(size_t)r*DDIM + tid]       : 0.f);
  float v1 = xr[tid + 256] + (R ? R[(size_t)r*DDIM + tid + 256] : 0.f);
  __shared__ float sm[256], sq[256];
  sm[tid] = v0 + v1; sq[tid] = v0*v0 + v1*v1; __syncthreads();
  for (int off = 128; off > 0; off >>= 1) {
    if (tid < off) { sm[tid] += sm[tid+off]; sq[tid] += sq[tid+off]; }
    __syncthreads();
  }
  __shared__ float mu, inv;
  if (tid == 0) {
    mu = sm[0] * (1.f/DDIM);
    float var = sq[0] * (1.f/DDIM) - mu*mu;
    inv = rsqrtf(var + 1e-5f);
  }
  __syncthreads();
  size_t orr = r;
  if (remap) { int b = r % BB, t = r / BB; orr = (size_t)b*LSEQ + t; }
  out[orr*DDIM + tid]       = (v0 - mu)*inv*g[tid]       + bt[tid];
  out[orr*DDIM + tid + 256] = (v1 - mu)*inv*g[tid + 256] + bt[tid + 256];
}

// ---------------------------------------------------------------------------

extern "C" void kernel_launch(void* const* d_in, const int* in_sizes, int n_in,
                              void* d_out, int out_size, void* d_ws, size_t ws_size,
                              hipStream_t stream)
{
  (void)in_sizes; (void)n_in; (void)out_size; (void)ws_size;
  const float* src_q = (const float*)d_in[0];
  const float* src_v = (const float*)d_in[1];
  // mamba: 0=m1a,1=m1v,2=m2a,3=m2v; fields: in_w,conv_w,conv_b,xproj_w,dt_w,dt_b,A_log,Dp,out_w
  auto Mp  = [&](int inst, int off){ return (const float*)d_in[2 + inst*9 + off]; };
  // dw: 0=dw1,1=dw2; fields: hp1_w,hp1_b,ep1_w,ep1_b,hp2_w,hp2_b,ep2_w,ep2_b
  auto DWp = [&](int idx, int off){ return (const float*)d_in[38 + idx*8 + off]; };
  // han: in_w,in_b,out_w,out_b,l1_w,l1_b,l2_w,l2_b,ln1_g,ln1_b,ln2_g,ln2_b
  auto HANp = [&](int off){ return (const float*)d_in[54 + off]; };

  float* wsp = (float*)d_ws;
  size_t pos = 0;
  auto alloc = [&](size_t n){ float* p = wsp + pos; pos += (n + 255) & ~(size_t)255; return p; };

  float* Qf = alloc(NROWS*DDIM);
  float* Vf = alloc(NROWS*DDIM);
  float *XZ[4], *XC[4], *XDB[4], *DT[4], *An[4];
  for (int i = 0; i < 4; ++i) {
    XZ[i]  = alloc((size_t)NROWS*2*DII);
    XC[i]  = alloc((size_t)NROWS*DII);
    XDB[i] = alloc((size_t)NROWS*160);
    DT[i]  = alloc((size_t)NROWS*DII);
    An[i]  = alloc((size_t)DII*DSS);
  }
  float *Ea[2], *Ev[2], *EB[4], *WT1[2], *WT2[2], *Yab[2], *Oa[2];
  for (int i = 0; i < 2; ++i) {
    Ea[i]  = alloc((size_t)NROWS*DII);
    Ev[i]  = alloc((size_t)NROWS*DII);
    WT1[i] = alloc((size_t)DII*DDIM);
    WT2[i] = alloc((size_t)DII*DDIM);
    Yab[i] = alloc((size_t)NROWS*DII);
    Oa[i]  = alloc((size_t)NROWS*DDIM);
  }
  for (int i = 0; i < 4; ++i) EB[i] = alloc(NROWS);
  float* QKVb = alloc((size_t)NROWS*3*DDIM);
  float* ATTb = alloc((size_t)NROWS*DDIM);
  float* S2b  = alloc((size_t)NROWS*DDIM);
  float* X1b  = alloc((size_t)NROWS*DDIM);
  float* XNb  = alloc((size_t)NROWS*DDIM);
  float* H1b  = alloc((size_t)NROWS*DDIM);
  float* F2b  = alloc((size_t)NROWS*DDIM);

  auto gemm = [&](const float* A, const float* W, const float* bias, float* C,
                  int M, int N, int K, int act){
    int tilesN = N/16;
    if ((tilesN & 3) == 0) {
      long grp = (long)(M/16) * (tilesN/4);
      k_gemm_wmma<4><<<(int)((grp + 7)/8), 256, 0, stream>>>(A, W, bias, C, M, N, K, act);
    } else if ((tilesN & 1) == 0) {
      long grp = (long)(M/16) * (tilesN/2);
      k_gemm_wmma<2><<<(int)((grp + 7)/8), 256, 0, stream>>>(A, W, bias, C, M, N, K, act);
    } else {
      long grp = (long)(M/16) * tilesN;
      k_gemm_wmma<1><<<(int)((grp + 7)/8), 256, 0, stream>>>(A, W, bias, C, M, N, K, act);
    }
  };
  int gEW = (NROWS*DII + 255)/256;
  int gED = (NROWS*DDIM + 255)/256;

  // ---- hoisted precompute ----
  k_transpose_in<<<gED, 256, 0, stream>>>(src_q, src_v, Qf, Vf);
  for (int i = 0; i < 4; ++i) {
    const float* src = (i == 1 || i == 3) ? Vf : Qf;
    int rev = (i >= 2);
    gemm(src, Mp(i,0), nullptr, XZ[i], NROWS, 2*DII, DDIM, 0);                   // in-proj
    k_conv_silu<<<gEW, 256, 0, stream>>>(XZ[i], Mp(i,1), Mp(i,2), XC[i], rev);   // conv+silu
    gemm(XC[i], Mp(i,3), nullptr, XDB[i], NROWS, 160, DII, 0);                   // x-proj (tilesN=10 -> NT=2)
    k_dt<<<gEW, 256, 0, stream>>>(XDB[i], Mp(i,4), Mp(i,5), DT[i]);              // softplus dt
    k_negexp<<<(DII*DSS + 255)/256, 256, 0, stream>>>(Mp(i,6), An[i], DII*DSS);  // A = -exp(A_log)
  }
  dim3 tgrid(DDIM/32, DII/32);
  for (int d = 0; d < 2; ++d) {
    k_transpose_w<<<tgrid, 256, 0, stream>>>(DWp(d,2), WT1[d], DDIM, DII);  // ep1_w^T
    k_transpose_w<<<tgrid, 256, 0, stream>>>(DWp(d,6), WT2[d], DDIM, DII);  // ep2_w^T
    gemm(Qf, WT1[d], nullptr, Ea[d], NROWS, DII, DDIM, 0);                  // Q @ ep1_w
    gemm(Vf, WT2[d], nullptr, Ev[d], NROWS, DII, DDIM, 0);                  // V @ ep2_w
    k_rowdot<<<2, 256, 0, stream>>>(Qf, DWp(d,3), EB[d*2+0]);
    k_rowdot<<<2, 256, 0, stream>>>(Vf, DWp(d,7), EB[d*2+1]);
  }

  // ---- the whole 60-step bidirectional scan: ONE launch ----
  ScanArgs sa[2];
  for (int d = 0; d < 2; ++d) {
    int ia = d*2, iv = d*2 + 1;
    sa[d].xconv_a = XC[ia]; sa[d].dt_a = DT[ia]; sa[d].xdb_a = XDB[ia];
    sa[d].A_a = An[ia]; sa[d].Dp_a = Mp(ia,7); sa[d].xz_a = XZ[ia];
    sa[d].xconv_v = XC[iv]; sa[d].dt_v = DT[iv]; sa[d].xdb_v = XDB[iv]; sa[d].A_v = An[iv];
    sa[d].Ea = Ea[d]; sa[d].Ev = Ev[d]; sa[d].eba = EB[d*2+0]; sa[d].ebv = EB[d*2+1];
    sa[d].hp1_w = DWp(d,0); sa[d].hp1_b = DWp(d,1);
    sa[d].hp2_w = DWp(d,4); sa[d].hp2_b = DWp(d,5);
    sa[d].Ya = Yab[d]; sa[d].rev = d;
  }
  k_scan<<<16, 1024, 0, stream>>>(sa[0], sa[1]);

  // ---- out-projections (only a-branch outputs are consumed) ----
  gemm(Yab[0], Mp(0,8), nullptr, Oa[0], NROWS, DDIM, DII, 0);
  gemm(Yab[1], Mp(2,8), nullptr, Oa[1], NROWS, DDIM, DII, 0);

  // ---- MHA ----
  gemm(Qf, HANp(0), HANp(1), QKVb, NROWS, 3*DDIM, DDIM, 0);
  k_mha<<<BB*8, 256, 0, stream>>>(QKVb, ATTb);
  gemm(ATTb, HANp(2), HANp(3), S2b, NROWS, DDIM, DDIM, 0);

  // ---- combine + LN + FFN + LN ----
  k_combine<<<gED, 256, 0, stream>>>(Qf, Oa[0], Oa[1], S2b, X1b);
  k_ln<<<NROWS, 256, 0, stream>>>(X1b, nullptr, HANp(8), HANp(9), XNb, 0);
  gemm(XNb, HANp(4), HANp(5), H1b, NROWS, DDIM, DDIM, 1);   // relu(x@l1^T+b)
  gemm(H1b, HANp(6), HANp(7), F2b, NROWS, DDIM, DDIM, 0);
  k_ln<<<NROWS, 256, 0, stream>>>(XNb, F2b, HANp(10), HANp(11), (float*)d_out, 1);
}
